// GemmaAttention_1005022347382
// MI455X (gfx1250) — compile-verified, matrix operand order checked by reading, not run on hardware
//
#include <hip/hip_runtime.h>
#include <hip/hip_bf16.h>

// ---------------------------------------------------------------------------
// GemmaAttention forward for MI455X (gfx1250, wave32, WMMA).
//   B=2, S=2048, D=2048, H=8, KV=1 (MQA), HD=256, fp32 I/O.
// Pipeline: f16 convert -> async-LDS block-tiled WMMA GEMM projections ->
// RoPE -> flash attention (online softmax, WMMA QK^T / PV) -> WMMA out proj.
// ---------------------------------------------------------------------------

#define NBATCH 2
#define SEQ    2048
#define DMODEL 2048
#define NHEAD  8
#define HEADD  256
#define QDIM   (NHEAD * HEADD)   // 2048

#define LDSPAD 40                // padded row stride (halves) for LDS tiles

typedef __attribute__((ext_vector_type(16))) _Float16 v16h;
typedef __attribute__((ext_vector_type(8)))  float    v8f;

union FragHalf {         // 16x32 (A) or 32x16 (B) f16 fragment: 16 halves/lane
  v16h v;
  uint4 q[2];
};

__device__ __forceinline__ v8f wmma_f32_f16(v16h a, v16h b, v8f c) {
  // (neg_a, A, neg_b, B, c_mod, C, reuse_a, reuse_b)
  return __builtin_amdgcn_wmma_f32_16x16x32_f16(false, a, false, b, (short)0, c,
                                                false, false);
}

// Per the CDNA5 ISA 16-bit A/B layout: lanes 0-15 hold k in {0..7,16..23},
// lanes 16-31 hold k in {8..15,24..31}.  koff = (lane>=16) ? 8 : 0.
// Each fragment = two contiguous 16-byte chunks per lane.
__device__ __forceinline__ FragHalf load_frag(const _Float16* rowptr, int koff) {
  FragHalf f;
  f.q[0] = *(const uint4*)(rowptr + koff);
  f.q[1] = *(const uint4*)(rowptr + koff + 16);
  return f;
}

// Async-engine copy: one 16B lane-transfer, global -> LDS (ASYNCcnt tracked).
__device__ __forceinline__ void async_copy_b128(unsigned lds_byte_off,
                                                const _Float16* g) {
  asm volatile("global_load_async_to_lds_b128 %0, %1, off"
               :: "v"(lds_byte_off), "v"((unsigned long long)(uintptr_t)g)
               : "memory");
}

__device__ __forceinline__ void wait_async_all() {
  asm volatile("s_wait_asynccnt 0" ::: "memory");
}

// ---------------------------------------------------------------------------
// Elementwise converts
// ---------------------------------------------------------------------------
__global__ __launch_bounds__(256)
void cvt_f32_f16_kernel(const float* __restrict__ in, _Float16* __restrict__ out,
                        int n) {
  int i = blockIdx.x * blockDim.x + threadIdx.x;
  if (i < n) out[i] = (_Float16)in[i];
}

// W: K x N fp32 (row-major)  ->  Wt: N x K f16 (row-major)
__global__ __launch_bounds__(256)
void transpose_cvt_kernel(const float* __restrict__ W, _Float16* __restrict__ Wt,
                          int K, int N) {
  int k = blockIdx.x * blockDim.x + threadIdx.x;
  int n = blockIdx.y;
  if (k < K) Wt[(size_t)n * K + k] = (_Float16)W[(size_t)k * N + n];
}

// RoPE (matches reference: inv_freq = base^(-2d/HD), rotate-half pairing
// (d, d+HD/2)) fused with f16 convert.  X: (B*S) x (nheads*HD) fp32.
__global__ __launch_bounds__(128)
void rope_cvt_kernel(const float* __restrict__ X, _Float16* __restrict__ Xh,
                     int nheads) {
  const int m = blockIdx.x;          // 0 .. B*S-1
  const int h = blockIdx.y;          // 0 .. nheads-1
  const int d = threadIdx.x;         // 0 .. 127
  const int pos = m & (SEQ - 1);
  const float inv_freq = __powf(10000.0f, -(float)(2 * d) / (float)HEADD);
  const float ang = (float)pos * inv_freq;
  float sn, cs;
  __sincosf(ang, &sn, &cs);
  const float* xp = X + (size_t)m * nheads * HEADD + (size_t)h * HEADD;
  _Float16*   yp = Xh + (size_t)m * nheads * HEADD + (size_t)h * HEADD;
  const float x0 = xp[d];
  const float x1 = xp[d + HEADD / 2];
  yp[d]             = (_Float16)(x0 * cs - x1 * sn);
  yp[d + HEADD / 2] = (_Float16)(x1 * cs + x0 * sn);
}

// V: (B*S) x HD fp32  ->  Vt: B x HD x S f16 (so PV B-fragments are contiguous)
__global__ __launch_bounds__(256)
void vtrans_cvt_kernel(const float* __restrict__ V, _Float16* __restrict__ Vt) {
  int s = blockIdx.x * blockDim.x + threadIdx.x;
  int d = blockIdx.y;
  int b = blockIdx.z;
  Vt[((size_t)b * HEADD + d) * SEQ + s] =
      (_Float16)V[((size_t)b * SEQ + s) * HEADD + d];
}

// ---------------------------------------------------------------------------
// Block-tiled WMMA GEMM with async-LDS double buffering:
//   C[M x N] (fp32) = A[M x K] (f16) @ Bt[N x K]^T (f16)
// Block = 256 threads (8 waves, 4m x 2n).  Block tile 128(M) x 128(N),
// k-step 32.  A/B 128x32 panels staged to LDS by the async engine
// (global_load_async_to_lds_b128, ASYNCcnt), double buffered.  Each wave
// computes 32x64 (2 m-subtiles x 4 n-subtiles, 8 f32 accumulators).
// All LDS fragment loads are hoisted ahead of the WMMA chain so the
// scheduler can cover DScnt latency once instead of per-WMMA.
// Requires M%128==0, N%128==0, K%32==0 (true for all uses here).
// ---------------------------------------------------------------------------
__global__ __launch_bounds__(256)
void gemm_xw_wmma(const _Float16* __restrict__ A, const _Float16* __restrict__ Bt,
                  float* __restrict__ C, int M, int N, int K) {
  // [buf][A=0/B=1][128 rows * LDSPAD halves]
  __shared__ _Float16 smem[2][2][128 * LDSPAD];

  const int tid = threadIdx.x;
  const int lane = tid & 31;
  const int wave = tid >> 5;
  const int nblocks = N >> 7;
  const int bm = blockIdx.x / nblocks;
  const int bn = blockIdx.x % nblocks;
  const int m_blk = bm << 7;
  const int n_blk = bn << 7;

  const int wave_m = wave >> 1;          // 0..3 -> 32 rows each
  const int wave_n = wave & 1;           // 0..1 -> 64 cols each
  const int halfsel = lane >> 4;
  const int lr = lane & 15;
  const int koff = halfsel << 3;

  const _Float16* agbase = A + (size_t)m_blk * K;
  const _Float16* bgbase = Bt + (size_t)n_blk * K;

  // Stage the 128x32 A and B panels into LDS buffer `buf` via async engine.
  auto stage = [&](int buf, int k0) {
    const unsigned abase = (unsigned)(uintptr_t)&smem[buf][0][0];
    const unsigned bbase = (unsigned)(uintptr_t)&smem[buf][1][0];
#pragma unroll
    for (int i = 0; i < 2; ++i) {
      const int idx = tid + i * 256;     // 0..511
      const int row = idx >> 2;          // 0..127
      const int chunk = idx & 3;         // 4 x 8-half chunks per row
      const unsigned loff = (unsigned)(row * LDSPAD + chunk * 8) * 2u;
      async_copy_b128(abase + loff, agbase + (size_t)row * K + k0 + chunk * 8);
      async_copy_b128(bbase + loff, bgbase + (size_t)row * K + k0 + chunk * 8);
    }
  };

  v8f acc[2][4];
#pragma unroll
  for (int mm = 0; mm < 2; ++mm)
#pragma unroll
    for (int nt = 0; nt < 4; ++nt) acc[mm][nt] = (v8f){};

  stage(0, 0);
  for (int k0 = 0; k0 < K; k0 += 32) {
    const int buf = (k0 >> 5) & 1;
    wait_async_all();
    __syncthreads();                     // panel visible to all waves
    if (k0 + 32 < K) stage(buf ^ 1, k0 + 32);

    const _Float16* Al = &smem[buf][0][0];
    const _Float16* Bl = &smem[buf][1][0];

    // Hoist all 12 ds_load_b128 ahead of the 8-WMMA chain.
    FragHalf af[2], bf[4];
#pragma unroll
    for (int mm = 0; mm < 2; ++mm)
      af[mm] = load_frag(Al + (wave_m * 32 + mm * 16 + lr) * LDSPAD, koff);
#pragma unroll
    for (int nt = 0; nt < 4; ++nt)
      bf[nt] = load_frag(Bl + (wave_n * 64 + nt * 16 + lr) * LDSPAD, koff);

#pragma unroll
    for (int nt = 0; nt < 4; ++nt)
#pragma unroll
      for (int mm = 0; mm < 2; ++mm)
        acc[mm][nt] = wmma_f32_f16(af[mm].v, bf[nt].v, acc[mm][nt]);

    __syncthreads();                     // all reads done before buf reused
  }

  // C/D layout: VGPR r -> lanes 0-15: (M=r, N=lane); lanes 16-31: (M=r+8)
#pragma unroll
  for (int mm = 0; mm < 2; ++mm) {
#pragma unroll
    for (int r = 0; r < 8; ++r) {
      const int row = m_blk + wave_m * 32 + mm * 16 + r + (halfsel << 3);
      float* crow = C + (size_t)row * N + n_blk + wave_n * 64 + lr;
#pragma unroll
      for (int nt = 0; nt < 4; ++nt) crow[nt * 16] = acc[mm][nt][r];
    }
  }
}

// ---------------------------------------------------------------------------
// Flash attention: one wave handles (b, h, 16 q-rows) across full HD=256.
// 32 keys per step: 16 QK^T WMMAs + online softmax + 16 PV WMMAs.
// P is re-laid-out C->A through a per-wave LDS staging buffer.
// ---------------------------------------------------------------------------
__global__ __launch_bounds__(256)
void flash_attn_wmma(const _Float16* __restrict__ Qh,  // (B*S) x QDIM
                     const _Float16* __restrict__ Kh,  // (B*S) x HD
                     const _Float16* __restrict__ Vt,  // B x HD x S
                     _Float16* __restrict__ Oh) {      // (B*S) x QDIM
  __shared__ _Float16 pbuf[8][16 * 32];                // per-wave P staging

  const int lane = threadIdx.x & 31;
  const int wave = threadIdx.x >> 5;
  const int gw = blockIdx.x * 8 + wave;
  const int qtiles = SEQ >> 4;                         // 128
  if (gw >= NBATCH * NHEAD * qtiles) return;
  const int qt = gw % qtiles;
  const int bh = gw / qtiles;
  const int h = bh % NHEAD;
  const int b = bh / NHEAD;
  const int q0 = qt << 4;

  const int halfsel = lane >> 4;
  const int lr = lane & 15;
  const int koff = halfsel << 3;

  // Q A-fragments: 8 chunks of k=32 over HD=256
  v16h qfrag[8];
  {
    const _Float16* qrow = Qh + (size_t)(b * SEQ + q0 + lr) * QDIM + (size_t)h * HEADD;
#pragma unroll
    for (int c = 0; c < 8; ++c) qfrag[c] = load_frag(qrow + c * 32, koff).v;
  }

  v8f acc[16];
#pragma unroll
  for (int t = 0; t < 16; ++t) acc[t] = (v8f){};
  float mmax[8], lsum[8];
#pragma unroll
  for (int r = 0; r < 8; ++r) { mmax[r] = -1e30f; lsum[r] = 0.0f; }

  const float scale = 0.0625f;                         // 1/sqrt(256)
  const _Float16* vbase = Vt + (size_t)b * HEADD * SEQ;
  const int nk = q0 + 16;                              // causal key count

  for (int kv0 = 0; kv0 < nk; kv0 += 32) {
    // prefetch next step's K rows
    if (kv0 + 32 < nk)
      __builtin_prefetch(Kh + (size_t)(b * SEQ + kv0 + 32 + lr) * HEADD, 0, 3);

    // ---- scores: two 16x16 tiles over 32 keys ----
    v8f st[2];
    st[0] = (v8f){};
    st[1] = (v8f){};
#pragma unroll
    for (int t = 0; t < 2; ++t) {
      const _Float16* krow = Kh + (size_t)(b * SEQ + kv0 + t * 16 + lr) * HEADD;
#pragma unroll
      for (int c = 0; c < 8; ++c) {
        FragHalf kb = load_frag(krow + c * 32, koff);
        st[t] = wmma_f32_f16(qfrag[c], kb.v, st[t]);
      }
    }
    // ---- scale + causal mask (elementwise, no divergence) ----
#pragma unroll
    for (int t = 0; t < 2; ++t) {
#pragma unroll
      for (int r = 0; r < 8; ++r) {
        const int row = q0 + r + (halfsel << 3);
        const int col = kv0 + t * 16 + lr;
        const float sv = st[t][r] * scale;
        st[t][r] = (col <= row) ? sv : -1e30f;
      }
    }
    // ---- online softmax row stats (16-lane halves via shfl_xor) ----
    float corr[8];
#pragma unroll
    for (int r = 0; r < 8; ++r) {
      float mx = fmaxf(st[0][r], st[1][r]);
      mx = fmaxf(mx, __shfl_xor(mx, 1, 32));
      mx = fmaxf(mx, __shfl_xor(mx, 2, 32));
      mx = fmaxf(mx, __shfl_xor(mx, 4, 32));
      mx = fmaxf(mx, __shfl_xor(mx, 8, 32));
      const float mnew = fmaxf(mmax[r], mx);
      corr[r] = __expf(mmax[r] - mnew);
      mmax[r] = mnew;
      const float p0 = __expf(st[0][r] - mnew);
      const float p1 = __expf(st[1][r] - mnew);
      st[0][r] = p0;
      st[1][r] = p1;
      float rs = p0 + p1;
      rs += __shfl_xor(rs, 1, 32);
      rs += __shfl_xor(rs, 2, 32);
      rs += __shfl_xor(rs, 4, 32);
      rs += __shfl_xor(rs, 8, 32);
      lsum[r] = lsum[r] * corr[r] + rs;
    }
    // ---- P: C layout -> A layout through per-wave LDS ----
    _Float16* pb = &pbuf[wave][0];
#pragma unroll
    for (int t = 0; t < 2; ++t) {
#pragma unroll
      for (int r = 0; r < 8; ++r) {
        const int prow = r + (halfsel << 3);
        const int pcol = t * 16 + lr;
        pb[prow * 32 + pcol] = (_Float16)st[t][r];
      }
    }
    asm volatile("s_wait_dscnt 0" ::: "memory");   // per-wave LDS RAW fence
    FragHalf pf;
    pf.q[0] = *(const uint4*)(pb + lr * 32 + koff);
    pf.q[1] = *(const uint4*)(pb + lr * 32 + koff + 16);

    // ---- rescale accumulators, then PV WMMAs ----
#pragma unroll
    for (int dt = 0; dt < 16; ++dt) {
#pragma unroll
      for (int r = 0; r < 8; ++r) acc[dt][r] *= corr[r];
    }
#pragma unroll
    for (int dt = 0; dt < 16; ++dt) {
      const _Float16* vrow = vbase + (size_t)(dt * 16 + lr) * SEQ + kv0;
      FragHalf vf = load_frag(vrow, koff);
      acc[dt] = wmma_f32_f16(pf.v, vf.v, acc[dt]);
    }
  }

  // ---- epilogue: normalize and write f16 attn output ----
  float linv[8];
#pragma unroll
  for (int r = 0; r < 8; ++r) linv[r] = 1.0f / lsum[r];
#pragma unroll
  for (int dt = 0; dt < 16; ++dt) {
#pragma unroll
    for (int r = 0; r < 8; ++r) {
      const int row = q0 + r + (halfsel << 3);
      const int col = dt * 16 + lr;
      Oh[(size_t)(b * SEQ + row) * QDIM + (size_t)h * HEADD + col] =
          (_Float16)(acc[dt][r] * linv[r]);
    }
  }
}

// ---------------------------------------------------------------------------
// Host-side launcher
// ---------------------------------------------------------------------------
extern "C" void kernel_launch(void* const* d_in, const int* in_sizes, int n_in,
                              void* d_out, int out_size, void* d_ws, size_t ws_size,
                              hipStream_t stream) {
  (void)in_sizes; (void)n_in; (void)out_size; (void)ws_size;
  const float* hidden = (const float*)d_in[0];
  // d_in[1] = attention_mask (causal; reproduced analytically)
  // d_in[2] = position_ids   (arange; reproduced analytically)
  const float* Wq = (const float*)d_in[3];
  const float* Wk = (const float*)d_in[4];
  const float* Wv = (const float*)d_in[5];
  const float* Wo = (const float*)d_in[6];
  float* out = (float*)d_out;

  const int M = NBATCH * SEQ;                    // 4096
  char* ws = (char*)d_ws;
  size_t off = 0;
  auto alloc = [&](size_t bytes) -> void* {
    void* p = ws + off;
    off += (bytes + 255) & ~(size_t)255;
    return p;
  };
  _Float16* hidden_h = (_Float16*)alloc((size_t)M * DMODEL * 2);
  _Float16* WqT = (_Float16*)alloc((size_t)QDIM * DMODEL * 2);
  _Float16* WkT = (_Float16*)alloc((size_t)HEADD * DMODEL * 2);
  _Float16* WvT = (_Float16*)alloc((size_t)HEADD * DMODEL * 2);
  _Float16* WoT = (_Float16*)alloc((size_t)DMODEL * QDIM * 2);
  float* Qf = (float*)alloc((size_t)M * QDIM * 4);
  float* Kf = (float*)alloc((size_t)M * HEADD * 4);
  float* Vf = (float*)alloc((size_t)M * HEADD * 4);
  _Float16* Qhh = (_Float16*)alloc((size_t)M * QDIM * 2);
  _Float16* Khh = (_Float16*)alloc((size_t)M * HEADD * 2);
  _Float16* Vth = (_Float16*)alloc((size_t)NBATCH * HEADD * SEQ * 2);
  _Float16* Ah  = (_Float16*)alloc((size_t)M * QDIM * 2);

  // 1) converts
  {
    int n = M * DMODEL;
    cvt_f32_f16_kernel<<<(n + 255) / 256, 256, 0, stream>>>(hidden, hidden_h, n);
  }
  transpose_cvt_kernel<<<dim3(DMODEL / 256, QDIM), 256, 0, stream>>>(Wq, WqT, DMODEL, QDIM);
  transpose_cvt_kernel<<<dim3(DMODEL / 256, HEADD), 256, 0, stream>>>(Wk, WkT, DMODEL, HEADD);
  transpose_cvt_kernel<<<dim3(DMODEL / 256, HEADD), 256, 0, stream>>>(Wv, WvT, DMODEL, HEADD);
  transpose_cvt_kernel<<<dim3(QDIM / 256, DMODEL), 256, 0, stream>>>(Wo, WoT, QDIM, DMODEL);

  // 2) projections (async-LDS block-tiled WMMA GEMMs, 128x128 block tiles)
  gemm_xw_wmma<<<(M / 128) * (QDIM / 128), 256, 0, stream>>>(hidden_h, WqT, Qf, M, QDIM, DMODEL);
  gemm_xw_wmma<<<(M / 128) * (HEADD / 128), 256, 0, stream>>>(hidden_h, WkT, Kf, M, HEADD, DMODEL);
  gemm_xw_wmma<<<(M / 128) * (HEADD / 128), 256, 0, stream>>>(hidden_h, WvT, Vf, M, HEADD, DMODEL);

  // 3) RoPE + f16 convert; V transpose-convert
  rope_cvt_kernel<<<dim3(M, NHEAD), 128, 0, stream>>>(Qf, Qhh, NHEAD);
  rope_cvt_kernel<<<dim3(M, 1), 128, 0, stream>>>(Kf, Khh, 1);
  vtrans_cvt_kernel<<<dim3(SEQ / 256, HEADD, NBATCH), 256, 0, stream>>>(Vf, Vth);

  // 4) flash attention
  {
    int waves = NBATCH * NHEAD * (SEQ >> 4);     // 2048
    flash_attn_wmma<<<(waves + 7) / 8, 256, 0, stream>>>(Qhh, Khh, Vth, Ah);
  }

  // 5) output projection -> fp32 result
  gemm_xw_wmma<<<(M / 128) * (DMODEL / 128), 256, 0, stream>>>(Ah, WoT, out, M, DMODEL, DMODEL);
}
